// FlexHNN_9156870275617
// MI455X (gfx1250) — compile-verified, matrix operand order checked by reading, not run on hardware
//
#include <hip/hip_runtime.h>
#include <hip/hip_bf16.h>

typedef __attribute__((ext_vector_type(16))) _Float16 v16h;
typedef __attribute__((ext_vector_type(8)))  _Float16 v8h;
typedef __attribute__((ext_vector_type(8)))  float    v8f;
typedef __attribute__((ext_vector_type(4)))  float    v4f;

#define NN      131072
#define HID     256
#define DZ      16
#define ROWS_PER_WAVE 16
#define WAVES_PER_BLK 4
#define ROWS_PER_BLK  (ROWS_PER_WAVE * WAVES_PER_BLK)

// ws layout (units = _Float16 halves). Each fragment = 32 lanes * 16 halves = 512 halves.
#define OFF_HB1 0u        // hW1 : JT=16, KT=1  -> 16 frags  (8192 halves)
#define OFF_HB2 8192u     // hW2 : JT=16, KT=8  -> 128 frags (65536 halves)
#define OFF_FB1 73728u    // fW1 : JT=16, KT=1  -> 16 frags
#define OFF_FB2 81920u    // fW2 : JT=16, KT=8  -> 128 frags
#define OFF_FB3 147456u   // fW3 : JT=1,  KT=8  -> 8 frags

// Hardware transcendental tanh (V_TANH_F32, TRANS32 op on CDNA5).
__device__ __forceinline__ float htanh(float x) {
#if __has_builtin(__builtin_amdgcn_tanhf)
    return __builtin_amdgcn_tanhf(x);
#elif __has_builtin(__builtin_amdgcn_tanh_f32)
    return __builtin_amdgcn_tanh_f32(x);
#else
    float r;
    // TRANS op: 1 independent op (v_nop) required before result use; raw asm
    // bypasses compiler hazard tracking, so pad explicitly.
    asm("v_tanh_f32 %0, %1\n\tv_nop" : "=v"(r) : "v"(x));
    return r;
#endif
}

// ---------------------------------------------------------------------------
// Prep: convert fp32 weight W[O x I] (a = x @ W^T) into f16 WMMA B-fragments.
// B[k, n] = W[jt*16 + n, kt*32 + k], zero-padded when kt*32+k >= I.
// Fragment element order matches v_wmma_f32_16x16x32_f16 B layout:
//   lane L, half h:  K = (L>=16 ? 8:0) + (h<8 ? h : (h-8)+16)
// Stored as dst[((kt*JT + jt)*32 + L)*16 + h] so the main kernel does one
// contiguous 32-byte per-lane load per fragment.
// ---------------------------------------------------------------------------
__global__ void prep_frags(const float* __restrict__ W, int I, int JT,
                           _Float16* __restrict__ dst) {
    int f   = blockIdx.x;
    int kt  = f / JT;
    int jt  = f % JT;
    int tid = threadIdx.x;        // 0..511
    int L   = tid >> 4;           // lane 0..31
    int h   = tid & 15;           // half 0..15
    int n   = L & 15;
    int kb  = (L >= 16) ? 8 : 0;
    int k   = kb + ((h < 8) ? h : (h - 8) + 16);
    int kg  = kt * 32 + k;
    float v = (kg < I) ? W[(size_t)(jt * 16 + n) * I + kg] : 0.0f;
    dst[((size_t)(kt * JT + jt) * 32 + L) * 16 + h] = (_Float16)v;
}

// Load an A-fragment (16x32 f16) for row M from a row-major [16][256] f16 LDS
// tile. lanes 0-15: K = kt*32 + {0..7, 16..23}; lanes 16-31: +8.
__device__ __forceinline__ v16h load_afrag(const _Float16* buf, int M, int cb, int kt) {
    const _Float16* p = buf + M * HID + kt * 32 + cb;
    v8h lo = *(const v8h*)(p);
    v8h hi = *(const v8h*)(p + 16);
    v16h a;
#pragma unroll
    for (int i = 0; i < 8; ++i) { a[i] = lo[i]; a[i + 8] = hi[i]; }
    return a;
}

__device__ __forceinline__ v8f wmma_f16(v16h a, v16h b, v8f c) {
    return __builtin_amdgcn_wmma_f32_16x16x32_f16(false, a, false, b,
                                                  (short)0, c, false, false);
}

__global__ __launch_bounds__(128) void flexhnn_main(
    const float* __restrict__ z,
    const float* __restrict__ hb1, const float* __restrict__ hb2,
    const float* __restrict__ fb1, const float* __restrict__ fb2,
    const float* __restrict__ fb3,
    const _Float16* __restrict__ ws,
    float* __restrict__ out)
{
    // per-wave private double buffer: [wave][2][16*256] f16  -> 64 KB / block
    __shared__ _Float16 lds[WAVES_PER_BLK * 2 * ROWS_PER_WAVE * HID];

    const int lane = threadIdx.x & 31;
    const int wave = threadIdx.x >> 5;
    _Float16* t1buf = &lds[wave * 2 * ROWS_PER_WAVE * HID];
    _Float16* gbuf  = t1buf + ROWS_PER_WAVE * HID;

    const int row0 = (blockIdx.x * WAVES_PER_BLK + wave) * ROWS_PER_WAVE;
    const int M    = lane & 15;              // A-row / C-column index
    const int cb   = (lane >= 16) ? 8 : 0;   // K/row offset for upper half-wave

    const v16h* hB1 = (const v16h*)(ws + OFF_HB1);
    const v16h* hB2 = (const v16h*)(ws + OFF_HB2);
    const v16h* fB1 = (const v16h*)(ws + OFF_FB1);
    const v16h* fB2 = (const v16h*)(ws + OFF_FB2);
    const v16h* fB3 = (const v16h*)(ws + OFF_FB3);

    const v8f zero = {0.f, 0.f, 0.f, 0.f, 0.f, 0.f, 0.f, 0.f};

    // ---- A-fragment of z (16x16 real, padded to K=32 with zeros) ----------
    v16h zA;
#pragma unroll
    for (int i = 0; i < 16; ++i) zA[i] = (_Float16)0.0f;
    {
        const float* zrow = z + (size_t)(row0 + M) * DZ + cb;
        v4f z0 = *(const v4f*)(zrow);
        v4f z1 = *(const v4f*)(zrow + 4);
#pragma unroll
        for (int i = 0; i < 4; ++i) {
            zA[i]     = (_Float16)z0[i];
            zA[4 + i] = (_Float16)z1[i];
        }
    }

    // ---- Phase 1: t1 = tanh(z @ hW1^T + hb1) -> LDS (f16, row-major) ------
    for (int jt = 0; jt < 16; ++jt) {
        v16h b = hB1[(size_t)jt * 32 + lane];
        v8f  c = wmma_f16(zA, b, zero);
        float bias = hb1[jt * 16 + M];
#pragma unroll
        for (int r = 0; r < 8; ++r) {
            float t = htanh(c[r] + bias);
            t1buf[(r + cb) * HID + jt * 16 + M] = (_Float16)t;
        }
    }

    // ---- Phase 2: a2 = t1 @ hW2^T + hb2 ; s2 = 1 - tanh(a2)^2 -------------
    //      f1 = z @ fW1^T + fb1 ; g1 = 0.5*f1*s2 + 0.5*tanh(f1) -> LDS ------
    for (int jt = 0; jt < 16; ++jt) {
        v8f acc = zero;
#pragma unroll
        for (int kt = 0; kt < 8; ++kt) {
            v16h a = load_afrag(t1buf, M, cb, kt);
            v16h b = hB2[((size_t)kt * 16 + jt) * 32 + lane];
            acc = wmma_f16(a, b, acc);
        }
        v16h bf = fB1[(size_t)jt * 32 + lane];
        v8f  f1 = wmma_f16(zA, bf, zero);
        float b2  = hb2[jt * 16 + M];
        float bf1 = fb1[jt * 16 + M];
#pragma unroll
        for (int r = 0; r < 8; ++r) {
            float t2 = htanh(acc[r] + b2);
            float s2 = 1.0f - t2 * t2;
            float fv = f1[r] + bf1;
            float g  = 0.5f * fv * s2 + 0.5f * htanh(fv);
            gbuf[(r + cb) * HID + jt * 16 + M] = (_Float16)g;
        }
    }

    // ---- Phase 3: f2 = g1 @ fW2^T + fb2 ; s1 = 1 - t1^2 (reload from LDS);
    //      g2 = 0.5*f2*s1 + 0.5*tanh(f2) -> overwrite t1buf ------------------
    for (int jt = 0; jt < 16; ++jt) {
        v8f acc = zero;
#pragma unroll
        for (int kt = 0; kt < 8; ++kt) {
            v16h a = load_afrag(gbuf, M, cb, kt);
            v16h b = fB2[((size_t)kt * 16 + jt) * 32 + lane];
            acc = wmma_f16(a, b, acc);
        }
        float b2 = fb2[jt * 16 + M];
#pragma unroll
        for (int r = 0; r < 8; ++r) {
            int   idx = (r + cb) * HID + jt * 16 + M;
            float t1v = (float)t1buf[idx];
            float s1  = 1.0f - t1v * t1v;
            float fv  = acc[r] + b2;
            float g   = 0.5f * fv * s1 + 0.5f * htanh(fv);
            t1buf[idx] = (_Float16)g;   // same lane read-then-write, in-order
        }
    }

    // ---- Phase 4: f3 = g2 @ fW3^T + fb3 ; symplectic swap+negate ----------
    {
        v8f acc = zero;
#pragma unroll
        for (int kt = 0; kt < 8; ++kt) {
            v16h a = load_afrag(t1buf, M, cb, kt);
            v16h b = fB3[(size_t)kt * 32 + lane];
            acc = wmma_f16(a, b, acc);
        }
        float b3  = fb3[M];
        int   col = (M + 8) & 15;                 // dz/dt column this lane feeds
        float sgn = (M < 8) ? -1.0f : 1.0f;       // -dH/dq part negated
#pragma unroll
        for (int r = 0; r < 8; ++r) {
            out[(size_t)(row0 + r + cb) * DZ + col] = sgn * (acc[r] + b3);
        }
    }
}

extern "C" void kernel_launch(void* const* d_in, const int* in_sizes, int n_in,
                              void* d_out, int out_size, void* d_ws, size_t ws_size,
                              hipStream_t stream) {
    // setup_inputs order: t, z, hW1, hb1, hW2, hb2, fW1, fb1, fW2, fb2, fW3, fb3
    const float* zp  = (const float*)d_in[1];
    const float* hW1 = (const float*)d_in[2];
    const float* hb1 = (const float*)d_in[3];
    const float* hW2 = (const float*)d_in[4];
    const float* hb2 = (const float*)d_in[5];
    const float* fW1 = (const float*)d_in[6];
    const float* fb1 = (const float*)d_in[7];
    const float* fW2 = (const float*)d_in[8];
    const float* fb2 = (const float*)d_in[9];
    const float* fW3 = (const float*)d_in[10];
    const float* fb3 = (const float*)d_in[11];

    _Float16* ws = (_Float16*)d_ws;

    // Pre-swizzle weights into f16 WMMA B-fragments (weights ~1 MB; stays in L2).
    prep_frags<<<16,  512, 0, stream>>>(hW1, DZ,  16, ws + OFF_HB1);
    prep_frags<<<128, 512, 0, stream>>>(hW2, HID, 16, ws + OFF_HB2);
    prep_frags<<<16,  512, 0, stream>>>(fW1, DZ,  16, ws + OFF_FB1);
    prep_frags<<<128, 512, 0, stream>>>(fW2, HID, 16, ws + OFF_FB2);
    prep_frags<<<8,   512, 0, stream>>>(fW3, HID, 1,  ws + OFF_FB3);

    flexhnn_main<<<NN / ROWS_PER_BLK, WAVES_PER_BLK * 32, 0, stream>>>(
        zp, hb1, hb2, fb1, fb2, fb3, ws, (float*)d_out);
}
